// Gemma3Attention_9062380995151
// MI455X (gfx1250) — matrix operand order repaired and checked by reading, not compiled, not run on hardware
//
#include <hip/hip_runtime.h>
#include <hip/hip_bf16.h>

// ---------------- constants ----------------
constexpr int S_    = 4096;
constexpr int HID_  = 2560;
constexpr int NH_   = 8;
constexpr int NKV_  = 4;
constexpr int HD_   = 256;
constexpr int WIN_  = 1024;
constexpr int QKVN_ = (NH_ + 2 * NKV_) * HD_;   // 4096
constexpr int OK_   = NH_ * HD_;                // 2048 (o-proj K)
constexpr float SCALING_ = 0.0625f;             // 256^-0.5
constexpr float EPS_ = 1e-6f;

// LDS tile: 128 rows x 16 K-pair dwords, padded to 17 dwords (bank-conflict free)
constexpr unsigned TILE_BYTES_ = 128u * 17u * 4u;       // 8704
constexpr unsigned BUF_BYTES_  = 2u * TILE_BYTES_;      // A + B = 17408
// dynamic LDS total = 2 * BUF_BYTES_ (double buffered) = 34816

// ---------------- types ----------------
typedef __attribute__((ext_vector_type(16))) __bf16        v16bf;
typedef __attribute__((ext_vector_type(8)))  float         v8f;
typedef __attribute__((ext_vector_type(8)))  unsigned int  v8u;
typedef __attribute__((ext_vector_type(4)))  unsigned int  u32x4;
typedef __attribute__((ext_vector_type(8)))  int           i32x8;
typedef __attribute__((ext_vector_type(4)))  int           i32x4;

static __device__ __forceinline__ unsigned short f32_to_bf16(float f) {
  unsigned int u = __builtin_bit_cast(unsigned int, f);
  u += 0x7FFFu + ((u >> 16) & 1u);              // round-to-nearest-even
  return (unsigned short)(u >> 16);
}
static __device__ __forceinline__ unsigned int pack2f(float lo, float hi) {
  return (unsigned int)f32_to_bf16(lo) | ((unsigned int)f32_to_bf16(hi) << 16);
}
static __device__ __forceinline__ unsigned int pack2(unsigned short lo, unsigned short hi) {
  return (unsigned int)lo | ((unsigned int)hi << 16);
}
static __device__ __forceinline__ v8f wmma_bf16(v8u a, v8u b, v8f c) {
  return __builtin_amdgcn_wmma_f32_16x16x32_bf16(
      false, __builtin_bit_cast(v16bf, a),
      false, __builtin_bit_cast(v16bf, b),
      (short)0, c, false, false);
}

// ---------------------------------------------------------------------------
// TDM: async-load a 2D bf16 tile (tile_d0 elems x tile_d1 rows, row stride
// `stride0` elems) into LDS at byte offset `ldsoff`, inserting a 1-dword pad
// every 16 dwords (-> 17-dword LDS rows). Descriptor per CDNA5 ISA §8.
// ---------------------------------------------------------------------------
static __device__ __forceinline__ void tdm_load_2d(unsigned ldsoff,
                                                   const void* gptr,
                                                   unsigned tensor_d0,
                                                   unsigned tensor_d1,
                                                   unsigned tile_d0,
                                                   unsigned tile_d1,
                                                   unsigned long long stride0) {
  const unsigned long long ga = (unsigned long long)gptr;
  u32x4 g0;
  g0[0] = 1u;                                        // count=1, user mode
  g0[1] = ldsoff;                                    // lds_addr
  g0[2] = (unsigned)ga;                              // global_addr[31:0]
  g0[3] = (unsigned)((ga >> 32) & 0x01FFFFFFu) | (2u << 30);  // addr[56:32], type=2
  i32x8 g1;
  g1[0] = (int)((1u << 16)     // data_size = 2 bytes
              | (1u << 20)     // pad_enable
              | (3u << 22)     // pad_interval = 16 dwords
              | (0u << 25));   // pad_amount   = 1 dword
  g1[1] = (int)((tensor_d0 & 0xFFFFu) << 16);                    // dim0[15:0]
  g1[2] = (int)((tensor_d0 >> 16) | ((tensor_d1 & 0xFFFFu) << 16));
  g1[3] = (int)((tensor_d1 >> 16) | (tile_d0 << 16));
  g1[4] = (int)(tile_d1 & 0xFFFFu);                              // tile_dim1
  g1[5] = (int)(unsigned)(stride0 & 0xFFFFFFFFull);              // dim0 stride
  g1[6] = (int)(unsigned)((stride0 >> 32) & 0xFFFFull);
  g1[7] = 0;
  i32x4 z4 = {0, 0, 0, 0};
#if __clang_major__ >= 23
  i32x8 z8 = {0, 0, 0, 0, 0, 0, 0, 0};
  __builtin_amdgcn_tensor_load_to_lds(g0, g1, z4, z4, z8, 0);
#else
  __builtin_amdgcn_tensor_load_to_lds(g0, g1, z4, z4, 0);
#endif
}

// ---------------------------------------------------------------------------
// Kernel: bf16 GEMM, TDM-staged, double-buffered.  C[M,N] = A[M,K] * Bt[N,K]^T
// A: bf16 row-major M x K.  Bt: bf16 row-major N x K (pre-transposed weights).
// Block 256 = 8 waves; tile 128x128; wave computes 32x64 (2x4 WMMA tiles).
// Requires M%128==0, N%128==0, K%32==0.
// ---------------------------------------------------------------------------
__global__ __launch_bounds__(256) void gemm_bf16_tdm(
    const unsigned short* __restrict__ A, const unsigned short* __restrict__ Bt,
    float* __restrict__ C, int M, int N, int K) {
  extern __shared__ char smem[];                 // 2 x (A-tile + B-tile)

  const int t    = threadIdx.x;
  const int wave = t >> 5;
  const int lane = t & 31;
  const int half = lane >> 4;
  const int lan  = lane & 15;
  const int wm   = wave >> 1;                    // 0..3 : 32-row strip
  const int wn   = wave & 1;                     // 0..1 : 64-col strip
  const int m0   = blockIdx.y * 128;
  const int n0   = blockIdx.x * 128;
  const int nsteps = K >> 5;

  v8f acc[2][4] = {};

  // issue TDM for k-slab `k0` into buffer `buf` (wave 0: A, wave 1: B)
  auto stage = [&](int buf, int k0) {
    if (wave == 0)
      tdm_load_2d((unsigned)buf * BUF_BYTES_,
                  A + (size_t)m0 * K + k0, (unsigned)K, (unsigned)M,
                  32u, 128u, (unsigned long long)K);
    else if (wave == 1)
      tdm_load_2d((unsigned)buf * BUF_BYTES_ + TILE_BYTES_,
                  Bt + (size_t)n0 * K + k0, (unsigned)K, (unsigned)N,
                  32u, 128u, (unsigned long long)K);
  };

  stage(0, 0);                                   // prologue
  for (int st = 0; st < nsteps; ++st) {
    if (wave < 2) __builtin_amdgcn_s_wait_tensorcnt((short)0);
    __syncthreads();                             // current buffer ready, prev reads done
    if (st + 1 < nsteps) stage((st + 1) & 1, (st + 1) * 32);  // overlap next DMA

    const unsigned int* As =
        (const unsigned int*)(smem + (st & 1) * BUF_BYTES_);
    const unsigned int* Bs =
        (const unsigned int*)(smem + (st & 1) * BUF_BYTES_ + TILE_BYTES_);

    v8u a[2], b[4];
#pragma unroll
    for (int mt = 0; mt < 2; ++mt) {
      const int r = wm * 32 + mt * 16 + lan;
#pragma unroll
      for (int v = 0; v < 8; ++v) {
        const int cu = (v < 4) ? (half * 4 + v) : (8 + half * 4 + (v - 4));
        a[mt][v] = As[r * 17 + cu];
      }
    }
#pragma unroll
    for (int nt = 0; nt < 4; ++nt) {
      const int n = wn * 64 + nt * 16 + lan;
#pragma unroll
      for (int v = 0; v < 8; ++v)
        b[nt][v] = Bs[n * 17 + ((v < 4) ? (half * 4 + v) : (8 + half * 4 + (v - 4)))];
    }
#pragma unroll
    for (int mt = 0; mt < 2; ++mt)
#pragma unroll
      for (int nt = 0; nt < 4; ++nt)
        acc[mt][nt] = wmma_bf16(a[mt], b[nt], acc[mt][nt]);
  }

#pragma unroll
  for (int mt = 0; mt < 2; ++mt)
#pragma unroll
    for (int nt = 0; nt < 4; ++nt)
#pragma unroll
      for (int v = 0; v < 8; ++v) {
        const int row = m0 + wm * 32 + mt * 16 + v + 8 * half;
        const int col = n0 + wn * 64 + nt * 16 + lan;
        C[(size_t)row * N + col] = acc[mt][nt][v];
      }
}

// ---------------------------------------------------------------------------
// fp32 -> bf16 bulk convert (n % 4 == 0)
// ---------------------------------------------------------------------------
__global__ __launch_bounds__(256) void cvt_bf16(const float* __restrict__ src,
                                                unsigned short* __restrict__ dst,
                                                int n) {
  const int i = (blockIdx.x * 256 + threadIdx.x) * 4;
  if (i < n) {
    const float4 f = *(const float4*)(src + i);
    uint2 p;
    p.x = pack2f(f.x, f.y);
    p.y = pack2f(f.z, f.w);
    *(uint2*)(dst + i) = p;
  }
}

// ---------------------------------------------------------------------------
// Transpose + convert: src fp32 (K x N) -> dst bf16 (N x K).  K,N % 32 == 0.
// ---------------------------------------------------------------------------
__global__ __launch_bounds__(256) void transpose_bf16(const float* __restrict__ src,
                                                      unsigned short* __restrict__ dst,
                                                      int K, int N) {
  __shared__ float tl[32][33];
  const int bx = blockIdx.x * 32;                // N offset
  const int by = blockIdx.y * 32;                // K offset
  const int tx = threadIdx.x & 31, ty = threadIdx.x >> 5;
#pragma unroll
  for (int r = 0; r < 32; r += 8)
    tl[ty + r][tx] = src[(size_t)(by + ty + r) * N + bx + tx];
  __syncthreads();
#pragma unroll
  for (int r = 0; r < 32; r += 8)
    dst[(size_t)(bx + ty + r) * K + by + tx] = f32_to_bf16(tl[tx][ty + r]);
}

// ---------------------------------------------------------------------------
// Fused Gemma RMSNorm + RoPE for Q,K; fp32->bf16 convert for V.
// One block per (seq position, head-slot): slots 0..7 Q, 8..11 K, 12..15 V.
// ---------------------------------------------------------------------------
__global__ __launch_bounds__(256) void normrope(const float* __restrict__ qkv,
                                                const int* __restrict__ positions,
                                                const float* __restrict__ qw,
                                                const float* __restrict__ kw,
                                                unsigned short* __restrict__ qb,
                                                unsigned short* __restrict__ kb,
                                                unsigned short* __restrict__ vb) {
  const int s    = blockIdx.x >> 4;
  const int slot = blockIdx.x & 15;
  const int d    = threadIdx.x;
  const float* rowp = qkv + (size_t)s * QKVN_;

  float val;
  if (slot < 8)       val = rowp[slot * HD_ + d];
  else if (slot < 12) val = rowp[NH_ * HD_ + (slot - 8) * HD_ + d];
  else                val = rowp[(NH_ + NKV_) * HD_ + (slot - 12) * HD_ + d];

  if (slot >= 12) {
    vb[((size_t)s * NKV_ + (slot - 12)) * HD_ + d] = f32_to_bf16(val);
    return;
  }

  __shared__ float red[8];
  __shared__ float xn[HD_];

  float sq = val * val;
#pragma unroll
  for (int m = 1; m < 32; m <<= 1) sq += __shfl_xor(sq, m, 32);
  if ((threadIdx.x & 31) == 0) red[threadIdx.x >> 5] = sq;
  __syncthreads();
  float tot = 0.f;
#pragma unroll
  for (int i = 0; i < 8; ++i) tot += red[i];
  const float inv = rsqrtf(tot / (float)HD_ + EPS_);

  const float* w = (slot < 8) ? qw : kw;
  xn[d] = val * inv * (1.0f + w[d]);
  __syncthreads();

  const int   pos = positions[s];
  const int   d2  = d & 127;
  const float invf = __powf(10000.0f, -(float)d2 / 128.0f);
  const float fr   = (float)pos * invf;
  const float c = __cosf(fr), sn = __sinf(fr);
  const float out = (d < 128) ? (xn[d] * c - xn[d + 128] * sn)
                              : (xn[d] * c + xn[d - 128] * sn);
  if (slot < 8) qb[((size_t)s * NH_  + slot)      * HD_ + d] = f32_to_bf16(out);
  else          kb[((size_t)s * NKV_ + (slot - 8)) * HD_ + d] = f32_to_bf16(out);
}

// ---------------------------------------------------------------------------
// Sliding-window flash attention, bf16 WMMA, online softmax. Output bf16.
// Grid (S/64, NH); block 128 = 4 waves; each wave owns 16 query rows.
// ---------------------------------------------------------------------------
__global__ __launch_bounds__(128) void attn(const unsigned short* __restrict__ qb,
                                            const unsigned short* __restrict__ kb,
                                            const unsigned short* __restrict__ vb,
                                            unsigned short* __restrict__ attn_out) {
  const int wave = threadIdx.x >> 5;
  const int lane = threadIdx.x & 31;
  const int half = lane >> 4;
  const int lan  = lane & 15;
  const int h    = blockIdx.y;
  const int kvh  = h >> 1;                     // GQA group = 2
  const int q0   = blockIdx.x * 64 + wave * 16;

  __shared__ unsigned short plds[4][16 * 32];  // per-wave P tile (bf16)
  unsigned short* pw = &plds[wave][0];
  const unsigned int* pw32 = (const unsigned int*)pw;

  v8u aq[8];
  {
    const unsigned int* qp = (const unsigned int*)
        (qb + ((size_t)(q0 + lan) * NH_ + h) * HD_);
#pragma unroll
    for (int ks = 0; ks < 8; ++ks)
#pragma unroll
      for (int v = 0; v < 8; ++v) {
        const int kd = ks * 32 + ((v < 4) ? (half * 8 + 2 * v)
                                          : (16 + half * 8 + 2 * (v - 4)));
        aq[ks][v] = qp[kd >> 1];
      }
  }

  v8f   acc[16] = {};
  float mrun[8], lrun[8];
#pragma unroll
  for (int v = 0; v < 8; ++v) { mrun[v] = -1e30f; lrun[v] = 0.f; }

  int jstart = q0 - (WIN_ - 1);
  if (jstart < 0) jstart = 0;
  jstart &= ~31;

  for (int jb = jstart; jb <= q0 + 15; jb += 32) {
    v8f s0 = {}, s1 = {};
    const unsigned int* kp0 = (const unsigned int*)
        (kb + ((size_t)(jb + lan)      * NKV_ + kvh) * HD_);
    const unsigned int* kp1 = (const unsigned int*)
        (kb + ((size_t)(jb + 16 + lan) * NKV_ + kvh) * HD_);
#pragma unroll
    for (int ks = 0; ks < 8; ++ks) {
      v8u b0, b1;
#pragma unroll
      for (int v = 0; v < 8; ++v) {
        const int kd = ks * 32 + ((v < 4) ? (half * 8 + 2 * v)
                                          : (16 + half * 8 + 2 * (v - 4)));
        b0[v] = kp0[kd >> 1];
        b1[v] = kp1[kd >> 1];
      }
      s0 = wmma_bf16(aq[ks], b0, s0);
      s1 = wmma_bf16(aq[ks], b1, s1);
    }

    float scalef[8];
#pragma unroll
    for (int v = 0; v < 8; ++v) {
      const int i  = q0 + v + 8 * half;
      const int j0 = jb + lan, j1 = jb + 16 + lan;
      const bool ok0 = (j0 <= i) && (j0 > i - WIN_);
      const bool ok1 = (j1 <= i) && (j1 > i - WIN_);
      float e0 = ok0 ? s0[v] * SCALING_ : -1e30f;
      float e1 = ok1 ? s1[v] * SCALING_ : -1e30f;

      float mx = fmaxf(e0, e1);
#pragma unroll
      for (int m = 1; m < 16; m <<= 1) mx = fmaxf(mx, __shfl_xor(mx, m, 32));
      const float nm = fmaxf(mrun[v], mx);
      const float sc = __expf(mrun[v] - nm);
      const float p0 = __expf(e0 - nm);
      const float p1 = __expf(e1 - nm);
      pw[(v + 8 * half) * 32 + lan]      = f32_to_bf16(p0);
      pw[(v + 8 * half) * 32 + 16 + lan] = f32_to_bf16(p1);
      float rs = p0 + p1;
#pragma unroll
      for (int m = 1; m < 16; m <<= 1) rs += __shfl_xor(rs, m, 32);
      lrun[v] = lrun[v] * sc + rs;
      mrun[v] = nm;
      scalef[v] = sc;
    }
#pragma unroll
    for (int tt = 0; tt < 16; ++tt)
#pragma unroll
      for (int v = 0; v < 8; ++v) acc[tt][v] *= scalef[v];

    v8u ap;
#pragma unroll
    for (int v = 0; v < 8; ++v) {
      const int kk = (v < 4) ? (half * 8 + 2 * v) : (16 + half * 8 + 2 * (v - 4));
      ap[v] = pw32[(lan * 32 + kk) >> 1];      // wave-local LDS: DS ops in-order
    }
#pragma unroll
    for (int tt = 0; tt < 16; ++tt) {
      v8u bv;
#pragma unroll
      for (int v = 0; v < 8; ++v) {
        const int kk = (v < 4) ? (half * 8 + 2 * v) : (16 + half * 8 + 2 * (v - 4));
        const size_t b0 = ((size_t)(jb + kk)     * NKV_ + kvh) * HD_ + tt * 16 + lan;
        const size_t b1 = ((size_t)(jb + kk + 1) * NKV_ + kvh) * HD_ + tt * 16 + lan;
        bv[v] = pack2(vb[b0], vb[b1]);
      }
      acc[tt] = wmma_bf16(ap, bv, acc[tt]);
    }
  }

#pragma unroll
  for (int tt = 0; tt < 16; ++tt)
#pragma unroll
    for (int v = 0; v < 8; ++v) {
      const int r = q0 + v + 8 * half;
      attn_out[(size_t)r * OK_ + h * HD_ + tt * 16 + lan] =
          f32_to_bf16(acc[tt][v] / lrun[v]);
    }
}

// ---------------------------------------------------------------------------
// Launch: convert/transpose -> qkv GEMM (TDM) -> norm/rope -> attention
//         -> output GEMM (TDM)
// Workspace (~162 MB): qkv f32 | q/k/v bf16 | hidden bf16 | Wqkv^T bf16 |
//                      Wo^T bf16 | attn_out bf16
// ---------------------------------------------------------------------------
extern "C" void kernel_launch(void* const* d_in, const int* in_sizes, int n_in,
                              void* d_out, int out_size, void* d_ws, size_t ws_size,
                              hipStream_t stream) {
  const int*   positions = (const int*)d_in[0];
  const float* hidden    = (const float*)d_in[1];
  const float* qkv_w     = (const float*)d_in[2];
  const float* o_w       = (const float*)d_in[3];
  const float* q_norm_w  = (const float*)d_in[4];
  const float* k_norm_w  = (const float*)d_in[5];
  float*       out       = (float*)d_out;

  char* ws = (char*)d_ws;
  size_t off = 0;
  float* qkv_f32 = (float*)(ws + off);           off += (size_t)S_ * QKVN_ * 4;
  unsigned short* qbuf = (unsigned short*)(ws + off); off += (size_t)S_ * NH_  * HD_ * 2;
  unsigned short* kbuf = (unsigned short*)(ws + off); off += (size_t)S_ * NKV_ * HD_ * 2;
  unsigned short* vbuf = (unsigned short*)(ws + off); off += (size_t)S_ * NKV_ * HD_ * 2;
  unsigned short* hbf  = (unsigned short*)(ws + off); off += (size_t)S_ * HID_ * 2;
  unsigned short* wqkv_t = (unsigned short*)(ws + off); off += (size_t)QKVN_ * HID_ * 2;
  unsigned short* wo_t   = (unsigned short*)(ws + off); off += (size_t)HID_ * OK_ * 2;
  unsigned short* attn_bf = (unsigned short*)(ws + off);

  // 0) precision prep: hidden -> bf16; weights -> bf16 transposed (N x K)
  cvt_bf16<<<(S_ * HID_ / 4 + 255) / 256, 256, 0, stream>>>(hidden, hbf, S_ * HID_);
  transpose_bf16<<<dim3(QKVN_ / 32, HID_ / 32), 256, 0, stream>>>(qkv_w, wqkv_t,
                                                                  HID_, QKVN_);
  transpose_bf16<<<dim3(HID_ / 32, OK_ / 32), 256, 0, stream>>>(o_w, wo_t,
                                                                OK_, HID_);

  // 1) QKV projection (TDM-staged, double buffered)
  gemm_bf16_tdm<<<dim3(QKVN_ / 128, S_ / 128), 256, 2 * BUF_BYTES_, stream>>>(
      hbf, wqkv_t, qkv_f32, S_, QKVN_, HID_);

  // 2) RMSNorm + RoPE (Q,K) and bf16 convert (V)
  normrope<<<S_ * 16, 256, 0, stream>>>(qkv_f32, positions, q_norm_w, k_norm_w,
                                        qbuf, kbuf, vbuf);

  // 3) Sliding-window attention -> bf16
  attn<<<dim3(S_ / 64, NH_), 128, 0, stream>>>(qbuf, kbuf, vbuf, attn_bf);

  // 4) Output projection (TDM-staged)
  gemm_bf16_tdm<<<dim3(HID_ / 128, S_ / 128), 256, 2 * BUF_BYTES_, stream>>>(
      attn_bf, wo_t, out, S_, HID_, OK_);
}
